// RelativeMultiHeadAttention_1340029796447
// MI455X (gfx1250) — matrix operand order repaired and checked by reading, not compiled, not run on hardware
//
#include <hip/hip_runtime.h>

#define S_  256
#define M_  256
#define B_  8
#define D_  512
#define H_  8
#define HD_ 64
#define K_  512

typedef __bf16 bf16;
typedef __attribute__((ext_vector_type(16))) __bf16 v16bf;
typedef __attribute__((ext_vector_type(8)))  float v8f;

static __device__ inline v8f wmma_bf16(v16bf a, v16bf b, v8f c) {
  // D = A(16x32) * B(32x16) + C, f32 accumulate
  return __builtin_amdgcn_wmma_f32_16x16x32_bf16(false, a, false, b, (short)0, c, false, false);
}

// ---- fragment loaders (bf16 sources only) -----------------------------------
// A fragment 16(M)x32(K): lane l<16 -> row rowBase+l, K = kBase + half*16 + e (contiguous 16)
static __device__ inline v16bf frag_a_bf16(const bf16* src, int ld, int rowBase, int kBase, int lane) {
  int l16 = lane & 15, half = lane >> 4;
  const bf16* p = src + (size_t)(rowBase + l16) * ld + kBase;
  v16bf f;
#pragma unroll
  for (int e = 0; e < 8; ++e) f[e] = p[half * 16 + e];
#pragma unroll
  for (int e = 0; e < 8; ++e) f[8 + e] = p[half * 16 + 8 + e];
  return f;
}
// B fragment 32(K)x16(N), operand rows-along-K (NT gemm: C = X * W^T):
// lane l<16 -> column colBase+l (a row of W), K = kBase + half*16 + e (contiguous 16)
static __device__ inline v16bf frag_b_nt_bf16(const bf16* src, int ld, int colBase, int kBase, int lane) {
  int l16 = lane & 15, half = lane >> 4;
  const bf16* p = src + (size_t)(colBase + l16) * ld + kBase + half * 16;
  v16bf f;
#pragma unroll
  for (int e = 0; e < 16; ++e) f[e] = p[e];
  return f;
}
// B fragment from a KxN row-major source (NN gemm: C = P * V): element (k, n)
static __device__ inline v16bf frag_b_nn_bf16(const bf16* src, int ld, int kBase, int colBase, int lane) {
  int l16 = lane & 15, half = lane >> 4;
  v16bf f;
#pragma unroll
  for (int e = 0; e < 16; ++e) f[e] = src[(size_t)(kBase + half * 16 + e) * ld + colBase + l16];
  return f;
}

// ---- one-time fp32 -> bf16 conversion (float4 vectorized) -------------------
__global__ __launch_bounds__(256) void k_cvt(const float* __restrict__ src, bf16* __restrict__ dst, int n) {
  int i = (blockIdx.x * 256 + threadIdx.x) * 4;
  if (i < n) {
    float4 v = *(const float4*)(src + i);
    dst[i + 0] = (bf16)v.x;
    dst[i + 1] = (bf16)v.y;
    dst[i + 2] = (bf16)v.z;
    dst[i + 3] = (bf16)v.w;
  }
}

// ---- projection kernels (all-bf16 operands) ---------------------------------
// qu/qv[b,h,s,d] = x @ Wq^T (+u / +v). Wave tile = 16x32 (A-frag reused 2x).
__global__ __launch_bounds__(256) void k_proj_q(const bf16* __restrict__ xb, const bf16* __restrict__ Wq,
                                                const float* __restrict__ ub, const float* __restrict__ vbias,
                                                bf16* __restrict__ qu, bf16* __restrict__ qv) {
  int lane = threadIdx.x & 31, wid = threadIdx.x >> 5;
  int tile = blockIdx.x * 8 + wid;                    // 128 row-tiles x 16 col-pairs
  int tM = (tile >> 4) << 4, tN = (tile & 15) << 5;
  v8f acc0 = {}, acc1 = {};
  for (int kb = 0; kb < D_; kb += 32) {
    v16bf a  = frag_a_bf16(xb, D_, tM, kb, lane);
    v16bf b0 = frag_b_nt_bf16(Wq, D_, tN, kb, lane);
    v16bf b1 = frag_b_nt_bf16(Wq, D_, tN + 16, kb, lane);
    acc0 = wmma_bf16(a, b0, acc0);
    acc1 = wmma_bf16(a, b1, acc1);
  }
  int l16 = lane & 15, half = lane >> 4;
#pragma unroll
  for (int t = 0; t < 2; ++t) {
    v8f acc = t ? acc1 : acc0;
    int col = tN + t * 16 + l16, h = col >> 6, d = col & 63;
    float uu = ub[col], vv = vbias[col];
#pragma unroll
    for (int r = 0; r < 8; ++r) {
      int i = tM + half * 8 + r;                      // i = s*B + b
      int s = i >> 3, b = i & 7;
      size_t o = ((size_t)((b * H_ + h) * S_ + s)) * HD_ + d;
      qu[o] = (bf16)(acc[r] + uu);
      qv[o] = (bf16)(acc[r] + vv);
    }
  }
}

// kb/vb[b,h,k,d] = kv @ {Wk,Wv}^T. Wave tile 16x32, A-frag shared by 4 WMMAs.
__global__ __launch_bounds__(256) void k_proj_kv(const bf16* __restrict__ kvb,
                                                 const bf16* __restrict__ Wk, const bf16* __restrict__ Wv,
                                                 bf16* __restrict__ kbuf, bf16* __restrict__ vbuf) {
  int lane = threadIdx.x & 31, wid = threadIdx.x >> 5;
  int tile = blockIdx.x * 8 + wid;                    // 256 row-tiles x 16 col-pairs
  int tM = (tile >> 4) << 4, tN = (tile & 15) << 5;
  v8f ak0 = {}, ak1 = {}, av0 = {}, av1 = {};
  for (int kb = 0; kb < D_; kb += 32) {
    v16bf a   = frag_a_bf16(kvb, D_, tM, kb, lane);
    v16bf bk0 = frag_b_nt_bf16(Wk, D_, tN, kb, lane);
    v16bf bk1 = frag_b_nt_bf16(Wk, D_, tN + 16, kb, lane);
    v16bf bv0 = frag_b_nt_bf16(Wv, D_, tN, kb, lane);
    v16bf bv1 = frag_b_nt_bf16(Wv, D_, tN + 16, kb, lane);
    ak0 = wmma_bf16(a, bk0, ak0);
    ak1 = wmma_bf16(a, bk1, ak1);
    av0 = wmma_bf16(a, bv0, av0);
    av1 = wmma_bf16(a, bv1, av1);
  }
  int l16 = lane & 15, half = lane >> 4;
#pragma unroll
  for (int t = 0; t < 2; ++t) {
    v8f ak = t ? ak1 : ak0;
    v8f av = t ? av1 : av0;
    int col = tN + t * 16 + l16, h = col >> 6, d = col & 63;
#pragma unroll
    for (int r = 0; r < 8; ++r) {
      int i = tM + half * 8 + r;                      // i = kpos*B + b
      int kk = i >> 3, b = i & 7;
      size_t o = ((size_t)((b * H_ + h) * K_ + kk)) * HD_ + d;
      kbuf[o] = (bf16)ak[r];
      vbuf[o] = (bf16)av[r];
    }
  }
}

// rb[h,krel,d] = rel_pos @ Wr^T. Wave tile 16x32.
__global__ __launch_bounds__(256) void k_proj_r(const bf16* __restrict__ relb, const bf16* __restrict__ Wr,
                                                bf16* __restrict__ rb) {
  int lane = threadIdx.x & 31, wid = threadIdx.x >> 5;
  int tile = blockIdx.x * 8 + wid;                    // 32 row-tiles x 16 col-pairs
  int tM = (tile >> 4) << 4, tN = (tile & 15) << 5;
  v8f acc0 = {}, acc1 = {};
  for (int kb = 0; kb < D_; kb += 32) {
    v16bf a  = frag_a_bf16(relb, D_, tM, kb, lane);
    v16bf b0 = frag_b_nt_bf16(Wr, D_, tN, kb, lane);
    v16bf b1 = frag_b_nt_bf16(Wr, D_, tN + 16, kb, lane);
    acc0 = wmma_bf16(a, b0, acc0);
    acc1 = wmma_bf16(a, b1, acc1);
  }
  int l16 = lane & 15, half = lane >> 4;
#pragma unroll
  for (int t = 0; t < 2; ++t) {
    v8f acc = t ? acc1 : acc0;
    int col = tN + t * 16 + l16, h = col >> 6, d = col & 63;
#pragma unroll
    for (int r = 0; r < 8; ++r) {
      int krel = tM + half * 8 + r;
      rb[((size_t)(h * K_ + krel)) * HD_ + d] = (bf16)acc[r];
    }
  }
}

// ---- flash attention with relative band GEMM --------------------------------
#define QB 32
#define KT 32
__global__ __launch_bounds__(128) void k_attn(const bf16* __restrict__ qu, const bf16* __restrict__ qv,
                                              const bf16* __restrict__ kbuf, const bf16* __restrict__ vbuf,
                                              const bf16* __restrict__ rbuf, bf16* __restrict__ ybuf) {
  __shared__ __align__(16) bf16 sQu[QB][HD_], sQv[QB][HD_], sKt[KT][HD_], sVt[KT][HD_], sRb[64][HD_];
  __shared__ float sBD[QB][64];
  __shared__ float sSc[QB][KT];
  __shared__ __align__(16) bf16 sP[QB][KT];
  __shared__ float sAl[QB], sLf[QB];

  int qblk = blockIdx.x, h = blockIdx.y, b = blockIdx.z;
  int q0 = qblk * QB;
  int tid = threadIdx.x, lane = tid & 31, wid = tid >> 5;
  int l16 = lane & 15, half = lane >> 4;

  // ---- stage Q blocks, 16B vectorized (QB*HD = 256 uint4)
  const uint4* quP = (const uint4*)(qu + ((size_t)(b * H_ + h) * S_ + q0) * HD_);
  const uint4* qvP = (const uint4*)(qv + ((size_t)(b * H_ + h) * S_ + q0) * HD_);
#pragma unroll
  for (int t = tid; t < QB * HD_ / 8; t += 128) {
    ((uint4*)&sQu[0][0])[t] = quP[t];
    ((uint4*)&sQv[0][0])[t] = qvP[t];
  }
  const bf16* kP = kbuf + (size_t)(b * H_ + h) * K_ * HD_;
  const bf16* vP = vbuf + (size_t)(b * H_ + h) * K_ * HD_;
  const bf16* rP = rbuf + (size_t)h * K_ * HD_;

  v8f o0 = {}, o1 = {};               // O tiles (rows 0-15 / 16-31, cols wid*16..)
  float mrow = -1e30f, lrow = 0.f;    // owned by threads 0..31 (row = tid)

  int nkt = qblk + 9;                 // causal: last needed k-tile contains q0+31+256
  if (nkt > K_ / KT) nkt = K_ / KT;
  for (int it = 0; it < nkt; ++it) {
    int k0 = it * KT;
    int J0 = q0 + M_ - k0 - (KT - 1); // min rel index in this tile
    __syncthreads();                  // protect sVt/sP reads of previous iter
    // ---- stage K/V tiles (KT*HD = 256 uint4 each; 2 per thread)
    {
      const uint4* kT = (const uint4*)(kP + (size_t)k0 * HD_);
      const uint4* vT = (const uint4*)(vP + (size_t)k0 * HD_);
#pragma unroll
      for (int t = tid; t < KT * HD_ / 8; t += 128) {
        ((uint4*)&sKt[0][0])[t] = kT[t];
        ((uint4*)&sVt[0][0])[t] = vT[t];
      }
    }
    // ---- stage clipped R band (64 rows x 8 uint4/row = 512 uint4; 4 per thread)
#pragma unroll
    for (int t = tid; t < 64 * HD_ / 8; t += 128) {
      int jj = t >> 3, d8 = t & 7;
      int rr = J0 + jj;
      rr = rr < 0 ? 0 : (rr > K_ - 1 ? K_ - 1 : rr);
      ((uint4*)&sRb[0][0])[t] = *(const uint4*)(rP + (size_t)rr * HD_ + d8 * 8);
    }
    // prefetch next K/V tile into cache while this tile computes
    if (it + 1 < nkt) {
      __builtin_prefetch(kP + (size_t)(k0 + KT) * HD_ + tid * 16, 0, 0);
      __builtin_prefetch(vP + (size_t)(k0 + KT) * HD_ + tid * 16, 0, 0);
    }
    __syncthreads();
    // ---- BD band GEMM: (Qv 32x64) @ (Rband 64x64)^T -> sBD 32x64
    {
      v8f d0 = {}, d1 = {};
      for (int kb = 0; kb < HD_; kb += 32) {
        v16bf a0 = frag_a_bf16(&sQv[0][0], HD_, 0, kb, lane);
        v16bf a1 = frag_a_bf16(&sQv[0][0], HD_, 16, kb, lane);
        v16bf bb = frag_b_nt_bf16(&sRb[0][0], HD_, wid * 16, kb, lane);
        d0 = wmma_bf16(a0, bb, d0);
        d1 = wmma_bf16(a1, bb, d1);
      }
      int cj = wid * 16 + l16;
#pragma unroll
      for (int r = 0; r < 8; ++r) {
        sBD[half * 8 + r][cj] = d0[r];
        sBD[16 + half * 8 + r][cj] = d1[r];
      }
    }
    __syncthreads();
    // ---- AC GEMM: (Qu 32x64) @ (Kt 32x64)^T, fuse band gather + scale + mask
    {
      int tM = (wid >> 1) * 16, tN = (wid & 1) * 16;
      v8f acc = {};
      for (int kb = 0; kb < HD_; kb += 32) {
        v16bf a = frag_a_bf16(&sQu[0][0], HD_, tM, kb, lane);
        v16bf bb = frag_b_nt_bf16(&sKt[0][0], HD_, tN, kb, lane);
        acc = wmma_bf16(a, bb, acc);
      }
      int kcol = k0 + tN + l16;
#pragma unroll
      for (int r = 0; r < 8; ++r) {
        int qr = tM + half * 8 + r;
        int qg = q0 + qr;
        float sc;
        if (kcol > qg + M_) {
          sc = -1e30f;                              // causal mask
        } else {
          int jj = (qg + M_ - kcol) - J0;           // 0..62, inside band
          sc = (acc[r] + sBD[qr][jj]) * 0.125f;     // 1/sqrt(HD)
        }
        sSc[qr][tN + l16] = sc;
      }
    }
    __syncthreads();
    // ---- online softmax (threads 0..31 each own one row)
    if (tid < QB) {
      float tmax = -1e30f;
      for (int kk = 0; kk < KT; ++kk) tmax = fmaxf(tmax, sSc[tid][kk]);
      float mnew = fmaxf(mrow, tmax);
      float al = __expf(mrow - mnew);
      float psum = 0.f;
      for (int kk = 0; kk < KT; ++kk) {
        float p = __expf(sSc[tid][kk] - mnew);
        psum += p;
        sP[tid][kk] = (bf16)p;
      }
      lrow = lrow * al + psum;
      mrow = mnew;
      sAl[tid] = al;
    }
    __syncthreads();
    // ---- O rescale + accumulate: O += (P 32x32) @ (Vt 32x64)
    {
#pragma unroll
      for (int r = 0; r < 8; ++r) {
        o0[r] *= sAl[half * 8 + r];
        o1[r] *= sAl[16 + half * 8 + r];
      }
      v16bf a0 = frag_a_bf16(&sP[0][0], KT, 0, 0, lane);
      v16bf a1 = frag_a_bf16(&sP[0][0], KT, 16, 0, lane);
      v16bf bb = frag_b_nn_bf16(&sVt[0][0], HD_, 0, wid * 16, lane);
      o0 = wmma_bf16(a0, bb, o0);
      o1 = wmma_bf16(a1, bb, o1);
    }
  }
  __syncthreads();
  if (tid < QB) sLf[tid] = lrow;
  __syncthreads();
  // ---- normalize, write y in (S,B,D) layout as bf16 for output projection
  int dcol = wid * 16 + l16;
#pragma unroll
  for (int r = 0; r < 8; ++r) {
    int qr0 = half * 8 + r, qr1 = 16 + half * 8 + r;
    float inv0 = 1.f / sLf[qr0], inv1 = 1.f / sLf[qr1];
    size_t o0i = ((size_t)(q0 + qr0) * B_ + b) * D_ + h * HD_ + dcol;
    size_t o1i = ((size_t)(q0 + qr1) * B_ + b) * D_ + h * HD_ + dcol;
    ybuf[o0i] = (bf16)(o0[r] * inv0);
    ybuf[o1i] = (bf16)(o1[r] * inv1);
  }
}

// ---- output projection: out(2048x512 f32) = y(bf16) @ Wo^T (bf16) -----------
__global__ __launch_bounds__(256) void k_out(const bf16* __restrict__ yb, const bf16* __restrict__ Wo,
                                             float* __restrict__ out) {
  int lane = threadIdx.x & 31, wid = threadIdx.x >> 5;
  int tile = blockIdx.x * 8 + wid;                    // 128 row-tiles x 16 col-pairs
  int tM = (tile >> 4) << 4, tN = (tile & 15) << 5;
  v8f acc0 = {}, acc1 = {};
  for (int kb = 0; kb < D_; kb += 32) {
    v16bf a  = frag_a_bf16(yb, D_, tM, kb, lane);
    v16bf b0 = frag_b_nt_bf16(Wo, D_, tN, kb, lane);
    v16bf b1 = frag_b_nt_bf16(Wo, D_, tN + 16, kb, lane);
    acc0 = wmma_bf16(a, b0, acc0);
    acc1 = wmma_bf16(a, b1, acc1);
  }
  int l16 = lane & 15, half = lane >> 4;
#pragma unroll
  for (int t = 0; t < 2; ++t) {
    v8f acc = t ? acc1 : acc0;
    int col = tN + t * 16 + l16;
#pragma unroll
    for (int r = 0; r < 8; ++r) {
      int i = tM + half * 8 + r;
      out[(size_t)i * D_ + col] = acc[r];
    }
  }
}

extern "C" void kernel_launch(void* const* d_in, const int* in_sizes, int n_in,
                              void* d_out, int out_size, void* d_ws, size_t ws_size,
                              hipStream_t stream) {
  const float* x   = (const float*)d_in[0];
  const float* mem = (const float*)d_in[1];
  const float* rel = (const float*)d_in[2];
  // d_in[3] = attn_mask (bool): reproduced analytically (k > q + M), not read
  const float* Wq  = (const float*)d_in[4];
  const float* Wk  = (const float*)d_in[5];
  const float* Wv  = (const float*)d_in[6];
  const float* Wr  = (const float*)d_in[7];
  const float* Wo  = (const float*)d_in[8];
  const float* ub  = (const float*)d_in[9];
  const float* vb  = (const float*)d_in[10];

  const size_t nKV = (size_t)(M_ + S_) * B_ * D_;       // 2M elems (kv concat rows x D)
  const size_t nX  = (size_t)S_ * B_ * D_;              // 1M
  const size_t nW  = (size_t)D_ * D_;                   // 256K
  const size_t nQ  = (size_t)B_ * H_ * S_ * HD_;        // 1M
  const size_t nK  = (size_t)B_ * H_ * K_ * HD_;        // 2M
  const size_t nR  = (size_t)H_ * K_ * HD_;             // 256K

  bf16* kvb  = (bf16*)d_ws;            // mem rows then x rows   4 MB
  bf16* relb = kvb + nKV;              //                        .5 MB
  bf16* wqb  = relb + nW;
  bf16* wkb  = wqb + nW;
  bf16* wvb  = wkb + nW;
  bf16* wrb  = wvb + nW;
  bf16* wob  = wrb + nW;               // 5 weights              2.5 MB
  bf16* qu   = wob + nW;               // (B,H,S,HD)             2 MB
  bf16* qv   = qu + nQ;                // (B,H,S,HD)             2 MB
  bf16* kb   = qv + nQ;                // (B,H,K,HD)             4 MB
  bf16* vv   = kb + nK;                // (B,H,K,HD)             4 MB
  bf16* rb   = vv + nK;                // (H,K,HD)               .5 MB
  bf16* yb   = rb + nR;                // (S,B,D)                2 MB

  // one-time conversions to bf16
  k_cvt<<<(int)(nX / 4 / 256), 256, 0, stream>>>(mem, kvb, (int)nX);        // kv rows 0..2047
  k_cvt<<<(int)(nX / 4 / 256), 256, 0, stream>>>(x, kvb + nX, (int)nX);     // kv rows 2048..4095
  k_cvt<<<(int)(nW / 4 / 256), 256, 0, stream>>>(rel, relb, (int)nW);
  k_cvt<<<(int)(nW / 4 / 256), 256, 0, stream>>>(Wq, wqb, (int)nW);
  k_cvt<<<(int)(nW / 4 / 256), 256, 0, stream>>>(Wk, wkb, (int)nW);
  k_cvt<<<(int)(nW / 4 / 256), 256, 0, stream>>>(Wv, wvb, (int)nW);
  k_cvt<<<(int)(nW / 4 / 256), 256, 0, stream>>>(Wr, wrb, (int)nW);
  k_cvt<<<(int)(nW / 4 / 256), 256, 0, stream>>>(Wo, wob, (int)nW);

  k_proj_q <<<256, 256, 0, stream>>>(kvb + nX, wqb, ub, vb, qu, qv);
  k_proj_kv<<<512, 256, 0, stream>>>(kvb, wkb, wvb, kb, vv);
  k_proj_r <<<64,  256, 0, stream>>>(relb, wrb, rb);
  k_attn   <<<dim3(S_ / QB, H_, B_), 128, 0, stream>>>(qu, qv, kb, vv, rb, yb);
  k_out    <<<256, 256, 0, stream>>>(yb, wob, (float*)d_out);
}
